// GAT_14491219657202
// MI455X (gfx1250) — matrix operand order repaired and checked by reading, not compiled, not run on hardware
//
#include <hip/hip_runtime.h>
#include <stdint.h>

// ---------------- problem constants (from reference) ----------------
#define N_NODES 50000
#define N_EDGES 800000
#define EP      (N_EDGES + N_NODES)   // edges + self loops = 850000
#define D_IN    256
#define H1      4
#define C1      64
#define HC1     256                   // H1*C1
#define C2      32
#define NEG_SLOPE 0.2f

typedef __attribute__((ext_vector_type(16))) _Float16 v16h;
typedef __attribute__((ext_vector_type(8)))  float    v8f;

// ---------- order-preserving float<->uint for atomic segment-max ----------
__device__ __forceinline__ unsigned f2ord(float f) {
    unsigned u = __float_as_uint(f);
    return (u & 0x80000000u) ? ~u : (u | 0x80000000u);
}
__device__ __forceinline__ float ord2f(unsigned u) {
    return __uint_as_float((u & 0x80000000u) ? (u ^ 0x80000000u) : ~u);
}
#define ORD_NEG_INF 0x007FFFFFu   // f2ord(-inf)

// ---------------- init kernels ----------------
__global__ void fill_f32(float* __restrict__ p, float v, int n) {
    int t = blockIdx.x * blockDim.x + threadIdx.x;
    if (t < n) p[t] = v;
}
__global__ void fill_u32(unsigned* __restrict__ p, unsigned v, int n) {
    int t = blockIdx.x * blockDim.x + threadIdx.x;
    if (t < n) p[t] = v;
}

// ---------------- weight pre-pack: f32 [K,Ncol] -> f16 B-fragments ----------
// Layout (v16h units): frag[colTile * (K/32) + ks][lane] , lane's 16 halves =
// B[ks*32 + kbB + j][colTile*16 + (lane&15)], kbB = (lane<16)?0:16.
// One thread per (colTile, ks, lane): 16 strided f32 reads -> 32B packed write.
__global__ void pack_w_k(const float* __restrict__ W, _Float16* __restrict__ Wp,
                         int K, int Ncol)
{
    int total = (Ncol / 16) * (K / 32) * 32;
    int t = blockIdx.x * blockDim.x + threadIdx.x;
    if (t >= total) return;
    int lane = t & 31;
    int rest = t >> 5;
    int nks  = K / 32;
    int ks   = rest % nks;
    int ct   = rest / nks;
    int n    = ct * 16 + (lane & 15);
    int kb   = ks * 32 + ((lane < 16) ? 0 : 16);
    _Float16* dst = Wp + (size_t)t * 16;
#pragma unroll
    for (int j = 0; j < 16; ++j)
        dst[j] = (_Float16)W[(size_t)(kb + j) * Ncol + n];
}

// ---------------- WMMA GEMM: D[M,NCOL] = A[M,K] @ W[K,NCOL] ----------------
// A: f32 in global, converted to f16 in registers (overlaps with XDL issue).
// B: pre-packed f16 fragments (one 32B v16h load per tile per K-step).
// Each wave owns a 16 x (16*NT) output strip: one A fragment feeds NT
// independent WMMAs per 32-deep K step. K/NCOL compile-time -> full unroll,
// immediate-offset addressing.
// VGPR layouts used here (A and B share contiguous k order, not the ISA table order):
//   A 16x32 f16 : lane(l): M = l&15, Kbase = (l<16)?0:16;
//                 halves j=0..15 -> K=Kbase+j
//   B 32x16 f16 : lane(l): N = l&15, Kbase = (l<16)?0:16; half j -> K=Kbase+j
//   C/D 16x16 f32: lane(l): N = l&15, vgpr v -> M = v + ((l<16)?0:8)
template<int K, int NCOL, int NT>
__global__ __launch_bounds__(128)
void gemm_f16_wmma(const float* __restrict__ A, const v16h* __restrict__ Wp,
                   float* __restrict__ Dst, int nColGroups, int nWaveTiles)
{
    int lane = threadIdx.x & 31;
    int wt   = blockIdx.x * (blockDim.x >> 5) + (threadIdx.x >> 5);
    if (wt >= nWaveTiles) return;               // wave-uniform exit

    int tm  = wt / nColGroups;                  // 16-row tile index
    int tg  = wt - tm * nColGroups;             // column-group index
    int hi  = lane >> 4;                        // 0 or 1 (lane half)
    int l15 = lane & 15;

    int rowA = tm * 16 + l15;
    int kbA  = hi * 16;
    int col0 = tg * (16 * NT) + l15;            // this lane's column in tile 0

    const int NKS = K / 32;

    v8f acc[NT] = {};
#pragma unroll
    for (int ks = 0; ks < NKS; ++ks) {
        // ---- A fragment (shared by all NT tiles): 4x float4 loads ----
        const float4* ap =
            (const float4*)(A + (size_t)rowA * K + (ks * 32 + kbA));
        float4 a0 = ap[0];          // K = kbA+0..3
        float4 a1 = ap[1];          // K = kbA+4..7
        float4 a2 = ap[2];          // K = kbA+8..11
        float4 a3 = ap[3];          // K = kbA+12..15
        v16h a;
        a[0]=(_Float16)a0.x; a[1]=(_Float16)a0.y; a[2]=(_Float16)a0.z; a[3]=(_Float16)a0.w;
        a[4]=(_Float16)a1.x; a[5]=(_Float16)a1.y; a[6]=(_Float16)a1.z; a[7]=(_Float16)a1.w;
        a[8]=(_Float16)a2.x; a[9]=(_Float16)a2.y; a[10]=(_Float16)a2.z; a[11]=(_Float16)a2.w;
        a[12]=(_Float16)a3.x; a[13]=(_Float16)a3.y; a[14]=(_Float16)a3.z; a[15]=(_Float16)a3.w;

#pragma unroll
        for (int t = 0; t < NT; ++t) {
            // packed B fragment: one 32B vector load, immediate offsets
            v16h b = Wp[((size_t)(tg * NT + t) * NKS + ks) * 32 + lane];
            acc[t] = __builtin_amdgcn_wmma_f32_16x16x32_f16(
                         false, a, false, b, (short)0, acc[t], false, false);
        }
    }

    int rbase = tm * 16 + hi * 8;
#pragma unroll
    for (int t = 0; t < NT; ++t)
#pragma unroll
        for (int v = 0; v < 8; ++v)
            Dst[(size_t)(rbase + v) * NCOL + (col0 + 16 * t)] = acc[t][v];
}

// ---------------- attention dot products: a_s/a_d = sum_c h*att ----------------
__global__ void att_dots(const float* __restrict__ h,
                         const float* __restrict__ att_s,
                         const float* __restrict__ att_d,
                         float* __restrict__ as_, float* __restrict__ ad_,
                         int n, int H, int C)
{
    int t = blockIdx.x * blockDim.x + threadIdx.x;
    if (t >= n * H) return;
    int node = t / H, hd = t - (t / H) * H;
    const float* hp = h + (size_t)node * H * C + hd * C;
    const float* sp = att_s + hd * C;
    const float* dp = att_d + hd * C;
    float s = 0.f, d = 0.f;
    for (int c = 0; c < C; ++c) { float v = hp[c]; s += v * sp[c]; d += v * dp[c]; }
    as_[t] = s; ad_[t] = d;
}

__device__ __forceinline__ void edge_sd(const int* __restrict__ ei, int e,
                                        int& s, int& d)
{
    if (e < N_EDGES) { s = ei[e]; d = ei[N_EDGES + e]; }
    else             { s = d = e - N_EDGES; }   // appended self-loop
}

__device__ __forceinline__ float leaky(float x) {
    return x > 0.f ? x : NEG_SLOPE * x;
}

// ---------------- edge pass 1: segment max over dst ----------------
__global__ void edge_max_k(const int* __restrict__ ei,
                           const float* __restrict__ as_,
                           const float* __restrict__ ad_,
                           unsigned* __restrict__ m, int H)
{
    int t = blockIdx.x * blockDim.x + threadIdx.x;
    if (t >= EP * H) return;
    int e = t / H, hd = t - e * H;
    int s, d; edge_sd(ei, e, s, d);
    float ev = leaky(as_[s * H + hd] + ad_[d * H + hd]);
    atomicMax(&m[d * H + hd], f2ord(ev));
}

// ---------------- edge pass 2: softmax denominator ----------------
__global__ void edge_den_k(const int* __restrict__ ei,
                           const float* __restrict__ as_,
                           const float* __restrict__ ad_,
                           const unsigned* __restrict__ m,
                           float* __restrict__ den, int H)
{
    int t = blockIdx.x * blockDim.x + threadIdx.x;
    if (t >= EP * H) return;
    int e = t / H, hd = t - e * H;
    int s, d; edge_sd(ei, e, s, d);
    float ev = leaky(as_[s * H + hd] + ad_[d * H + hd]);
    float p  = __expf(ev - ord2f(m[d * H + hd]));
    atomicAdd(&den[d * H + hd], p);
}

// ---------------- edge pass 3: weighted scatter-add of messages ----------------
// one thread per (edge, 4-channel group); agg[dst] += p * h[src]
__global__ void edge_agg_k(const int* __restrict__ ei,
                           const float* __restrict__ as_,
                           const float* __restrict__ ad_,
                           const unsigned* __restrict__ m,
                           const float* __restrict__ hsrc,
                           float* __restrict__ agg, int H, int C)
{
    int HC  = H * C;
    int gpe = HC >> 2;                       // 4-channel groups per edge
    int t = blockIdx.x * blockDim.x + threadIdx.x;
    if (t >= EP * gpe) return;
    int e = t / gpe, g = t - e * gpe;
    int ch0 = g << 2;
    int hd  = ch0 / C;
    int s, d; edge_sd(ei, e, s, d);
    float ev = leaky(as_[s * H + hd] + ad_[d * H + hd]);
    float p  = __expf(ev - ord2f(m[d * H + hd]));
    const float4 hv = *(const float4*)(hsrc + (size_t)s * HC + ch0);
    float* ap = agg + (size_t)d * HC + ch0;
    atomicAdd(ap + 0, p * hv.x);
    atomicAdd(ap + 1, p * hv.y);
    atomicAdd(ap + 2, p * hv.z);
    atomicAdd(ap + 3, p * hv.w);
}

// ---------------- layer-1 epilogue: normalize, +bias, ELU (in place) ----------------
__global__ void finish1_k(float* __restrict__ agg, const float* __restrict__ den,
                          const float* __restrict__ b, int n)
{
    int t = blockIdx.x * blockDim.x + threadIdx.x;
    if (t >= n * HC1) return;
    int node = t / HC1, ch = t - node * HC1;
    int hd = ch >> 6;                         // ch / C1
    float v = agg[t] / den[node * H1 + hd] + b[ch];
    agg[t] = v > 0.f ? v : (__expf(v) - 1.0f);
}

// ---------------- layer-2 epilogue: normalize + bias (in place on d_out) ----------------
__global__ void finish2_k(float* __restrict__ out, const float* __restrict__ den,
                          const float* __restrict__ b, int n)
{
    int t = blockIdx.x * blockDim.x + threadIdx.x;
    if (t >= n * C2) return;
    int node = t / C2, c = t - node * C2;
    out[t] = out[t] / den[node] + b[c];
}

// ---------------------------------------------------------------------------
extern "C" void kernel_launch(void* const* d_in, const int* in_sizes, int n_in,
                              void* d_out, int out_size, void* d_ws, size_t ws_size,
                              hipStream_t stream)
{
    (void)in_sizes; (void)n_in; (void)out_size; (void)ws_size;

    const float* x    = (const float*)d_in[0];
    const int*   ei   = (const int*)  d_in[1];
    const float* W1   = (const float*)d_in[2];
    const float* aS1  = (const float*)d_in[3];
    const float* aD1  = (const float*)d_in[4];
    const float* b1   = (const float*)d_in[5];
    const float* W2   = (const float*)d_in[6];
    const float* aS2  = (const float*)d_in[7];
    const float* aD2  = (const float*)d_in[8];
    const float* b2   = (const float*)d_in[9];
    float*       out  = (float*)d_out;

    // ---- workspace carve-up ----
    // Packed weights first (32B-aligned for v16h loads), then f32 arrays.
    _Float16* Wp1 = (_Float16*)d_ws;                       // 256*256 halves (128KB)
    _Float16* Wp2 = Wp1 + (size_t)D_IN * HC1;              // 256*32 halves (16KB)
    float*    h1   = (float*)(Wp2 + (size_t)HC1 * C2);     // [N, 256]
    float*    agg1 = h1   + (size_t)N_NODES * HC1;         // [N, 256] (becomes hact)
    float*    as1  = agg1 + (size_t)N_NODES * HC1;         // [N, 4]
    float*    ad1  = as1  + (size_t)N_NODES * H1;          // [N, 4]
    unsigned* m1   = (unsigned*)(ad1 + (size_t)N_NODES * H1);  // [N, 4]
    float*    den1 = (float*)(m1 + (size_t)N_NODES * H1);  // [N, 4]
    float*    h2   = den1 + (size_t)N_NODES * H1;          // [N, 32]
    float*    as2  = h2   + (size_t)N_NODES * C2;          // [N]
    float*    ad2  = as2  + N_NODES;                       // [N]
    unsigned* m2   = (unsigned*)(ad2 + N_NODES);           // [N]
    float*    den2 = (float*)(m2 + N_NODES);               // [N]

    auto cdiv = [](long long a, long long b) { return (int)((a + b - 1) / b); };
    const int TB = 256;

    // ---- re-initialize all accumulated state (deterministic replay) ----
    fill_f32<<<cdiv((long long)N_NODES * HC1, TB), TB, 0, stream>>>(agg1, 0.f, N_NODES * HC1);
    fill_f32<<<cdiv((long long)N_NODES * H1,  TB), TB, 0, stream>>>(den1, 0.f, N_NODES * H1);
    fill_u32<<<cdiv((long long)N_NODES * H1,  TB), TB, 0, stream>>>(m1, ORD_NEG_INF, N_NODES * H1);
    fill_f32<<<cdiv((long long)N_NODES * C2,  TB), TB, 0, stream>>>(out, 0.f, N_NODES * C2);
    fill_f32<<<cdiv((long long)N_NODES,       TB), TB, 0, stream>>>(den2, 0.f, N_NODES);
    fill_u32<<<cdiv((long long)N_NODES,       TB), TB, 0, stream>>>(m2, ORD_NEG_INF, N_NODES);

    // ---- pre-pack weights into WMMA B-fragment layout (f16) ----
    pack_w_k<<<cdiv((HC1 / 16) * (D_IN / 32) * 32, TB), TB, 0, stream>>>(W1, Wp1, D_IN, HC1);
    pack_w_k<<<cdiv((C2  / 16) * (HC1  / 32) * 32, TB), TB, 0, stream>>>(W2, Wp2, HC1, C2);

    // ================= layer 1 =================
    {   // h1 = x @ W1   (50000x256 @ 256x256), 16x64 strip per wave
        const int NT = 4;
        int nColGroups = (HC1 / 16) / NT;                // 4
        int nWaveTiles = (N_NODES / 16) * nColGroups;    // 12500
        gemm_f16_wmma<D_IN, HC1, NT><<<cdiv(nWaveTiles, 4), 128, 0, stream>>>(
            x, (const v16h*)Wp1, h1, nColGroups, nWaveTiles);
    }
    att_dots<<<cdiv((long long)N_NODES * H1, TB), TB, 0, stream>>>(h1, aS1, aD1, as1, ad1, N_NODES, H1, C1);
    edge_max_k<<<cdiv((long long)EP * H1, TB), TB, 0, stream>>>(ei, as1, ad1, m1, H1);
    edge_den_k<<<cdiv((long long)EP * H1, TB), TB, 0, stream>>>(ei, as1, ad1, m1, den1, H1);
    edge_agg_k<<<cdiv((long long)EP * (HC1 / 4), TB), TB, 0, stream>>>(ei, as1, ad1, m1, h1, agg1, H1, C1);
    finish1_k<<<cdiv((long long)N_NODES * HC1, TB), TB, 0, stream>>>(agg1, den1, b1, N_NODES);

    // ================= layer 2 =================
    {   // h2 = elu(agg1) @ W2   (50000x256 @ 256x32), 16x32 strip per wave
        const int NT = 2;
        int nColGroups = (C2 / 16) / NT;                 // 1
        int nWaveTiles = (N_NODES / 16) * nColGroups;    // 3125
        gemm_f16_wmma<HC1, C2, NT><<<cdiv(nWaveTiles, 4), 128, 0, stream>>>(
            agg1, (const v16h*)Wp2, h2, nColGroups, nWaveTiles);
    }
    att_dots<<<cdiv((long long)N_NODES, TB), TB, 0, stream>>>(h2, aS2, aD2, as2, ad2, N_NODES, 1, C2);
    edge_max_k<<<cdiv((long long)EP, TB), TB, 0, stream>>>(ei, as2, ad2, m2, 1);
    edge_den_k<<<cdiv((long long)EP, TB), TB, 0, stream>>>(ei, as2, ad2, m2, den2, 1);
    edge_agg_k<<<cdiv((long long)EP * (C2 / 4), TB), TB, 0, stream>>>(ei, as2, ad2, m2, h2, out, 1, C2);
    finish2_k<<<cdiv((long long)N_NODES * C2, TB), TB, 0, stream>>>(out, den2, b2, N_NODES);
}
